// SCNN_21260088115610
// MI455X (gfx1250) — matrix operand order repaired and checked by reading, not compiled, not run on hardware
//
#include <hip/hip_runtime.h>
#include <stdint.h>

// ---------------- constants ----------------
#define TW        25          // time window
#define NB        128         // batch
#define DECAY     0.8f
#define THRESH    0.7f
#define KEEP      0.6f        // 1 - 1/tau_s
#define INVTAU    0.4f        // 1/tau_s
#define BN_EPS    1e-5f

typedef __attribute__((ext_vector_type(16))) _Float16 v16h;
typedef __attribute__((ext_vector_type(8)))  _Float16 v8h;
typedef __attribute__((ext_vector_type(8)))  float    v8f;

#if defined(__has_builtin)
#if __has_builtin(__builtin_amdgcn_tensor_load_to_lds) && __has_builtin(__builtin_amdgcn_s_wait_tensorcnt)
#define HAVE_TDM 1
#endif
#endif
#ifndef HAVE_TDM
#define HAVE_TDM 0
#endif

#if HAVE_TDM
// 1-D TDM copy: global -> LDS, nbytes must be a multiple of 8 and /8 must fit 16 bits.
// D# per CDNA5 ISA 8.3/8.4: group0={count=1, lds_addr, global_addr, type=2};
// group1: data_size=8B, tensor_dim0=tile_dim0=nbytes/8, tensor_dim1=tile_dim1=1;
// groups 2/3 zero (<=2D tensor). Issued by one wave; tracked by TENSORcnt.
// This toolchain exposes the 6-arg builtin: (u32x4, i32x8, i32x4, i32x4, i32x8, i32 cpol).
__device__ __forceinline__ void tdm_copy_to_lds(const void* gsrc, void* lds_dst,
                                                unsigned nbytes) {
  typedef __attribute__((ext_vector_type(4))) unsigned int u32x4;
  typedef __attribute__((ext_vector_type(8))) int i32x8;
  typedef __attribute__((ext_vector_type(4))) int i32x4;
  unsigned long long ga = (unsigned long long)(uintptr_t)gsrc;
  unsigned lds0 = (unsigned)(uintptr_t)lds_dst;   // low 32 bits of generic ptr = LDS offset
  unsigned ne = nbytes >> 3;                      // 8-byte elements
  u32x4 g0;
  g0[0] = 1u;                                                   // count=1 (valid D#)
  g0[1] = lds0;                                                 // lds_addr
  g0[2] = (unsigned)ga;                                         // global_addr[31:0]
  g0[3] = (unsigned)((ga >> 32) & 0x01FFFFFFu) | (2u << 30);    // addr[56:32] | type=2
  i32x8 g1;
  g1[0] = (int)(3u << 16);                        // data_size=3 (8 bytes)
  g1[1] = (int)((ne & 0xFFFFu) << 16);            // tensor_dim0[15:0]
  g1[2] = (int)((ne >> 16) | (1u << 16));         // tensor_dim0[31:16] | tensor_dim1=1
  g1[3] = (int)((ne & 0xFFFFu) << 16);            // tile_dim0
  g1[4] = 1;                                      // tile_dim1=1, tile_dim2=0
  g1[5] = (int)ne;                                // tensor_dim0_stride
  g1[6] = 0;
  g1[7] = 0;
  i32x4 z4; z4[0] = 0; z4[1] = 0; z4[2] = 0; z4[3] = 0;
  i32x8 z8; z8[0] = 0; z8[1] = 0; z8[2] = 0; z8[3] = 0;
  z8[4] = 0; z8[5] = 0; z8[6] = 0; z8[7] = 0;
  __builtin_amdgcn_tensor_load_to_lds(g0, g1, z4, z4, z8, 0);
  __builtin_amdgcn_s_wait_tensorcnt(0);
}
#endif

// Stage nbytes of gsrc into LDS (TDM if available, cooperative copy otherwise),
// then barrier. Call from all threads of the block.
__device__ __forceinline__ void stage_to_lds(const _Float16* __restrict__ gsrc,
                                             _Float16* lds_dst, unsigned nbytes) {
#if HAVE_TDM
  if ((threadIdx.x >> 5) == 0) tdm_copy_to_lds(gsrc, lds_dst, nbytes);
#else
  unsigned nv = nbytes >> 4;  // 16B chunks
  for (unsigned i = threadIdx.x; i < nv; i += blockDim.x)
    ((v8h*)lds_dst)[i] = ((const v8h*)gsrc)[i];
#endif
  __syncthreads();
}

// Load a 16x32 f16 A-fragment from a row-major [M][ld] f16 matrix.
// Wave32 A layout (contiguous, matches packed B): lane<16 -> row=lane&15, K = kk+{0..15};
// lane>=16 -> same row, K = kk+{16..31}. Two contiguous 16B runs -> 2x b128.
__device__ __forceinline__ v16h load_a_frag(const _Float16* __restrict__ A,
                                            int row, int ld, int kk, int lane) {
  const _Float16* p = A + row * ld + kk + ((lane & 16) ? 16 : 0);
  v8h lo = *(const v8h*)p;
  v8h hi = *(const v8h*)(p + 8);
  v16h a;
#pragma unroll
  for (int i = 0; i < 8; ++i) { a[i] = lo[i]; a[i + 8] = hi[i]; }
  return a;
}

// ---------------- utility ----------------
__global__ void k_zero(float* __restrict__ p, int n) {
  int i = blockIdx.x * 256 + threadIdx.x;
  if (i < n) p[i] = 0.f;
}

// psp1 = (0.6*psp1 + x[..,t]) / tau ; x is [B,2,34,34,T]
__global__ void k_psp1(float* __restrict__ psp1, const float* __restrict__ x, int t) {
  int i = blockIdx.x * 256 + threadIdx.x;   // exact grid: 295936
  psp1[i] = (KEEP * psp1[i] + x[i * TW + t]) * INVTAU;
}

// ---------------- layer 1 ----------------
// im2col for conv1: A1h [M=131072][K=64], k=(ic*25+kh*5+kw), pad=1, K zero-padded to 64
__global__ void k_im2col1(const float* __restrict__ psp1, _Float16* __restrict__ A) {
  int idx = blockIdx.x * 256 + threadIdx.x;     // exact: 131072*64
  int m = idx >> 6, k = idx & 63;
  float v = 0.f;
  if (k < 50) {
    int b = m >> 10, rem = m & 1023, oh = rem >> 5, ow = rem & 31;
    int ic = k / 25, r = k - ic * 25, kh = r / 5, kw = r - kh * 5;
    int ih = oh - 1 + kh, iw = ow - 1 + kw;
    if (ih >= 0 && ih < 34 && iw >= 0 && iw < 34)
      v = psp1[((b * 2 + ic) * 34 + ih) * 34 + iw];
  }
  A[idx] = (_Float16)v;
}

// Pack conv1_w [12,2,5,5] into contiguous-k B fragments: VGPR v holds K=2v,2v+1 (lo/hi half);
// lanes 16-31 -> K+16; col = lane&15
__global__ void k_pack_b1(const float* __restrict__ w1, _Float16* __restrict__ Bp) {
  int idx = blockIdx.x * 256 + threadIdx.x;     // exact: 512
  int v = idx & 7, lane = (idx >> 3) & 31, ks = idx >> 8;
  int col = lane & 15;
  int k0 = ks * 32 + ((lane & 16) ? 16 : 0) + 2 * v;
  float f0 = (col < 12 && k0     < 50) ? w1[col * 50 + k0]     : 0.f;
  float f1 = (col < 12 && k0 + 1 < 50) ? w1[col * 50 + k0 + 1] : 0.f;
  _Float16* d = Bp + (ks * 32 + lane) * 16 + 2 * v;
  d[0] = (_Float16)f0; d[1] = (_Float16)f1;
}

// 16x16 tile GEMM, K=64 (2 WMMA k-steps). c1 layout [B,12,32,32]
__global__ void k_gemm1(const _Float16* __restrict__ A, const _Float16* __restrict__ Bp,
                        const float* __restrict__ bias, float* __restrict__ c1) {
  int lane = threadIdx.x & 31;
  int mtile = blockIdx.x * 8 + (threadIdx.x >> 5);  // exact: 1024 blk * 8 waves = 8192
  int row = mtile * 16 + (lane & 15);
  v8f acc = {};
#pragma unroll
  for (int ks = 0; ks < 2; ++ks) {
    v16h a = load_a_frag(A, row, 64, ks * 32, lane);
    v16h b = *(const v16h*)(Bp + (ks * 32 + lane) * 16);
    acc = __builtin_amdgcn_wmma_f32_16x16x32_f16(false, a, false, b, (short)0, acc,
                                                 false, false);
  }
  int col = lane & 15;
  if (col < 12) {
    float bb = bias[col];
    int mbase = mtile * 16 + ((lane & 16) ? 8 : 0);
#pragma unroll
    for (int r = 0; r < 8; ++r) {
      int m = mbase + r;
      int b_ = m >> 10, rem = m & 1023;
      c1[((b_ * 12 + col) << 10) + rem] = acc[r] + bb;
    }
  }
}

// Batch-norm statistics: stats[0..C-1]=sum, stats[C..2C-1]=sumsq (LDS partials)
__global__ void k_bnstats(const float* __restrict__ x, float* __restrict__ stats,
                          int total, int plane, int C) {
  __shared__ float ssum[64], ssq[64];
  for (int c = threadIdx.x; c < C; c += blockDim.x) { ssum[c] = 0.f; ssq[c] = 0.f; }
  __syncthreads();
  int stride = gridDim.x * blockDim.x;
  for (int i = blockIdx.x * blockDim.x + threadIdx.x; i < total; i += stride) {
    float v = x[i];
    int c = (i / plane) % C;
    atomicAdd(&ssum[c], v);
    atomicAdd(&ssq[c], v * v);
  }
  __syncthreads();
  for (int c = threadIdx.x; c < C; c += blockDim.x) {
    atomicAdd(&stats[c], ssum[c]);
    atomicAdd(&stats[C + c], ssq[c]);
  }
}

// fused BN + LIF + 2x2 maxpool + psp2 update; one thread per pooled output
__global__ void k_lif1pool(const float* __restrict__ c1, const float* __restrict__ stats,
                           float* __restrict__ mem1, float* __restrict__ psp2) {
  int idx = blockIdx.x * 256 + threadIdx.x;     // exact: 128*12*16*16 = 393216
  int b = idx / 3072, rem = idx % 3072;
  int c = rem >> 8, ph = (rem >> 4) & 15, pw = rem & 15;
  const float invN = 1.0f / 131072.0f;
  float mean = stats[c] * invN;
  float var  = stats[12 + c] * invN - mean * mean;
  float rs = rsqrtf(var + BN_EPS);
  float pooled = 0.f;
#pragma unroll
  for (int dy = 0; dy < 2; ++dy)
#pragma unroll
    for (int dx = 0; dx < 2; ++dx) {
      int oh = 2 * ph + dy, ow = 2 * pw + dx;
      int ci = ((b * 12 + c) * 32 + oh) * 32 + ow;
      float mthr = DECAY * mem1[ci] + (c1[ci] - mean) * rs;
      float spike = mthr > THRESH ? 1.f : 0.f;
      mem1[ci] = mthr - THRESH * spike;
      pooled = fmaxf(pooled, spike);
    }
  psp2[idx] = (KEEP * psp2[idx] + pooled) * INVTAU;
}

// ---------------- layer 2 ----------------
// im2col for conv2: A2h [M=18432][K=320], pad=0, input 16x16, K padded 300->320
__global__ void k_im2col2(const float* __restrict__ psp2, _Float16* __restrict__ A) {
  int idx = blockIdx.x * 256 + threadIdx.x;     // exact: 18432*320
  int m = idx / 320, k = idx - m * 320;
  float v = 0.f;
  if (k < 300) {
    int b = m / 144, rem = m - b * 144, oh = rem / 12, ow = rem - oh * 12;
    int ic = k / 25, r = k - ic * 25, kh = r / 5, kw = r - kh * 5;
    v = psp2[((b * 12 + ic) * 16 + oh + kh) * 16 + ow + kw];
  }
  A[idx] = (_Float16)v;
}

// Pack conv2_w [64,12,5,5] into [nt=4][ks=10][32 lanes] v16h fragments
__global__ void k_pack_b2(const float* __restrict__ w2, _Float16* __restrict__ Bp) {
  int idx = blockIdx.x * 256 + threadIdx.x;     // exact: 10240
  int v = idx & 7, lane = (idx >> 3) & 31, ks = (idx >> 8) % 10, nt = idx / 2560;
  int col = nt * 16 + (lane & 15);
  int k0 = ks * 32 + ((lane & 16) ? 16 : 0) + 2 * v;
  float f0 = (k0     < 300) ? w2[col * 300 + k0]     : 0.f;
  float f1 = (k0 + 1 < 300) ? w2[col * 300 + k0 + 1] : 0.f;
  _Float16* d = Bp + ((nt * 10 + ks) * 32 + lane) * 16 + 2 * v;
  d[0] = (_Float16)f0; d[1] = (_Float16)f1;
}

// GEMM2: M=18432, N=64, K=320 (10 WMMA k-steps). B fragments staged to LDS via TDM.
__global__ void k_gemm2(const _Float16* __restrict__ A, const _Float16* __restrict__ Bp,
                        const float* __restrict__ bias, float* __restrict__ c2) {
  __shared__ __align__(32) _Float16 bsm[20480];   // 4 ntiles * 10 ks * 32 lanes * 16 halves
  stage_to_lds(Bp, bsm, 20480u * 2u);
  int lane = threadIdx.x & 31;
  int w = blockIdx.x * 8 + (threadIdx.x >> 5);  // exact: 576 blk * 8 = 4608 waves
  int mtile = w >> 2, nt = w & 3;
  int row = mtile * 16 + (lane & 15);
  const _Float16* bpw = bsm + (nt * 10 * 32) * 16;
  v8f acc = {};
  for (int ks = 0; ks < 10; ++ks) {
    if (ks + 1 < 10)
      __builtin_prefetch(A + row * 320 + (ks + 1) * 32, 0, 3);  // global_prefetch_b8
    v16h a = load_a_frag(A, row, 320, ks * 32, lane);
    v16h b = *(const v16h*)(bpw + (ks * 32 + lane) * 16);       // ds_load_b128 x2
    acc = __builtin_amdgcn_wmma_f32_16x16x32_f16(false, a, false, b, (short)0, acc,
                                                 false, false);
  }
  int n = nt * 16 + (lane & 15);
  float bb = bias[n];
  int mbase = mtile * 16 + ((lane & 16) ? 8 : 0);
#pragma unroll
  for (int r = 0; r < 8; ++r) {
    int m = mbase + r;
    int b_ = m / 144, rem = m - b_ * 144;
    c2[(b_ * 64 + n) * 144 + rem] = acc[r] + bb;
  }
}

// fused BN + LIF + 2x2 maxpool -> flat f16 pre-scaled by 1/tau (psp3 = flat/tau)
__global__ void k_lif2pool(const float* __restrict__ c2, const float* __restrict__ stats,
                           float* __restrict__ mem2, _Float16* __restrict__ flath) {
  int idx = blockIdx.x * 256 + threadIdx.x;     // exact: 128*64*6*6 = 294912
  int b = idx / 2304, rem = idx % 2304;
  int c = rem / 36, r2 = rem % 36, ph = r2 / 6, pw = r2 % 6;
  const float invN = 1.0f / 18432.0f;
  float mean = stats[c] * invN;
  float var  = stats[64 + c] * invN - mean * mean;
  float rs = rsqrtf(var + BN_EPS);
  float pooled = 0.f;
#pragma unroll
  for (int dy = 0; dy < 2; ++dy)
#pragma unroll
    for (int dx = 0; dx < 2; ++dx) {
      int oh = 2 * ph + dy, ow = 2 * pw + dx;
      int ci = ((b * 64 + c) * 12 + oh) * 12 + ow;
      float mthr = DECAY * mem2[ci] + (c2[ci] - mean) * rs;
      float spike = mthr > THRESH ? 1.f : 0.f;
      mem2[ci] = mthr - THRESH * spike;
      pooled = fmaxf(pooled, spike);
    }
  flath[b * 2304 + c * 36 + ph * 6 + pw] = (_Float16)(pooled * INVTAU);
}

// ---------------- linear + output LIF ----------------
// Pack lin_w [10,2304] into [ks=72][32 lanes] v16h fragments (N padded to 16)
__global__ void k_pack_blin(const float* __restrict__ lw, _Float16* __restrict__ Bp) {
  int idx = blockIdx.x * 256 + threadIdx.x;     // exact: 18432
  int v = idx & 7, lane = (idx >> 3) & 31, ks = idx >> 8;
  int col = lane & 15;
  int k0 = ks * 32 + ((lane & 16) ? 16 : 0) + 2 * v;
  float f0 = (col < 10) ? lw[col * 2304 + k0]     : 0.f;
  float f1 = (col < 10) ? lw[col * 2304 + k0 + 1] : 0.f;
  _Float16* d = Bp + (ks * 32 + lane) * 16 + 2 * v;
  d[0] = (_Float16)f0; d[1] = (_Float16)f1;
}

// M=128 (8 waves), N=16, K=2304 (72 WMMA k-steps), fused with output LIF.
// 72KB of packed weights staged to LDS via TDM once per block.
__global__ void k_linlif(const _Float16* __restrict__ Aflat, const _Float16* __restrict__ Bp,
                         const float* __restrict__ lb, float* __restrict__ mem3,
                         float* __restrict__ out, int t) {
  __shared__ __align__(32) _Float16 bsm[36864];   // 72 ks * 32 lanes * 16 halves
  stage_to_lds(Bp, bsm, 36864u * 2u);
  int lane = threadIdx.x & 31;
  int mtile = threadIdx.x >> 5;                 // one block of 256 = 8 waves
  int row = mtile * 16 + (lane & 15);
  v8f acc = {};
  for (int ks = 0; ks < 72; ++ks) {
    v16h a = load_a_frag(Aflat, row, 2304, ks * 32, lane);
    v16h b = *(const v16h*)(bsm + (ks * 32 + lane) * 16);
    acc = __builtin_amdgcn_wmma_f32_16x16x32_f16(false, a, false, b, (short)0, acc,
                                                 false, false);
  }
  int n = lane & 15;
  if (n < 10) {
    float bb = lb[n];
    int mbase = mtile * 16 + ((lane & 16) ? 8 : 0);
#pragma unroll
    for (int r = 0; r < 8; ++r) {
      int m = mbase + r;
      float mthr = DECAY * mem3[m * 10 + n] + acc[r] + bb;
      float spike = mthr > THRESH ? 1.f : 0.f;
      mem3[m * 10 + n] = mthr - THRESH * spike;
      out[t * (NB * 10) + m * 10 + n] = spike;
    }
  }
}

// ---------------- host ----------------
extern "C" void kernel_launch(void* const* d_in, const int* in_sizes, int n_in,
                              void* d_out, int out_size, void* d_ws, size_t ws_size,
                              hipStream_t stream) {
  const float* x  = (const float*)d_in[0];   // [128,2,34,34,25]
  const float* w1 = (const float*)d_in[1];   // [12,2,5,5]
  const float* b1 = (const float*)d_in[2];   // [12]
  const float* w2 = (const float*)d_in[3];   // [64,12,5,5]
  const float* b2 = (const float*)d_in[4];   // [64]
  const float* lw = (const float*)d_in[5];   // [10,2304]
  const float* lb = (const float*)d_in[6];   // [10]
  float* out = (float*)d_out;                // [25,128,10]

  char* ws = (char*)d_ws;
  size_t off = 0;
  auto alloc = [&](size_t bytes) -> void* {
    off = (off + 255) & ~(size_t)255;
    void* p = ws + off;
    off += bytes;
    return p;
  };

  // state (zeroed each call; contiguous block)
  float* psp1  = (float*)alloc(295936u * 4);          // [B,2,34,34]
  float* mem1  = (float*)alloc(1572864u * 4);         // [B,12,32,32]
  float* psp2  = (float*)alloc(393216u * 4);          // [B,12,16,16]
  float* mem2  = (float*)alloc(1179648u * 4);         // [B,64,12,12]
  float* mem3  = (float*)alloc(1280u * 4);            // [B,10]
  float* stats = (float*)alloc(192u * 4);             // [0..23]=L1, [24..151]=L2
  // scratch
  float* c1buf = (float*)alloc(1572864u * 4);
  float* c2buf = (float*)alloc(1179648u * 4);
  _Float16* A1h   = (_Float16*)alloc(8388608u * 2);   // [131072][64]
  _Float16* A2h   = (_Float16*)alloc(5898240u * 2);   // [18432][320]
  _Float16* flath = (_Float16*)alloc(294912u * 2);    // [128][2304]
  _Float16* Bp1   = (_Float16*)alloc(1024u * 2);
  _Float16* Bp2   = (_Float16*)alloc(20480u * 2);
  _Float16* BpL   = (_Float16*)alloc(36864u * 2);

  // zero all persistent state in one shot (contiguous: psp1..stats)
  const int stateN = 295936 + 1572864 + 393216 + 1179648 + 1280 + 192;
  k_zero<<<(stateN + 255) / 256, 256, 0, stream>>>(psp1, stateN);

  // pack constant weights into WMMA B-fragment layouts (once per call)
  k_pack_b1<<<2, 256, 0, stream>>>(w1, Bp1);
  k_pack_b2<<<40, 256, 0, stream>>>(w2, Bp2);
  k_pack_blin<<<72, 256, 0, stream>>>(lw, BpL);

  float* stats2 = stats + 24;
  for (int t = 0; t < TW; ++t) {
    k_zero<<<1, 256, 0, stream>>>(stats, 192);
    k_psp1<<<1156, 256, 0, stream>>>(psp1, x, t);
    k_im2col1<<<32768, 256, 0, stream>>>(psp1, A1h);
    k_gemm1<<<1024, 256, 0, stream>>>(A1h, Bp1, b1, c1buf);
    k_bnstats<<<1024, 256, 0, stream>>>(c1buf, stats, 1572864, 1024, 12);
    k_lif1pool<<<1536, 256, 0, stream>>>(c1buf, stats, mem1, psp2);
    k_im2col2<<<23040, 256, 0, stream>>>(psp2, A2h);
    k_gemm2<<<576, 256, 0, stream>>>(A2h, Bp2, b2, c2buf);
    k_bnstats<<<1024, 256, 0, stream>>>(c2buf, stats2, 1179648, 144, 64);
    k_lif2pool<<<1152, 256, 0, stream>>>(c2buf, stats2, mem2, flath);
    k_linlif<<<1, 256, 0, stream>>>(flath, BpL, lb, mem3, out, t);
  }
}
